// QuadraticActiveDendrites_55035710931817
// MI455X (gfx1250) — matrix operand order repaired and checked by reading, not compiled, not run on hardware
//
#include <hip/hip_runtime.h>
#include <hip/hip_bf16.h>
#include <math.h>

// Problem sizes (fixed by the reference)
constexpr int BSZ  = 8192;   // batch
constexpr int INSZ = 2048;   // input features (K)
constexpr int OUTS = 2048;   // output features (N)
constexpr int DCTX = 1024;   // context dim
constexpr int NSEG = 16;     // segments per neuron

// ---- CDNA5 WMMA types ------------------------------------------------------
typedef __attribute__((ext_vector_type(16))) __bf16       v16bf;
typedef __attribute__((ext_vector_type(2)))  __bf16       v2bf;
typedef __attribute__((ext_vector_type(8)))  float        v8f;
typedef __attribute__((ext_vector_type(4)))  unsigned int v4u;

struct V4x2 { v4u a, b; };

// round-to-nearest-even f32 -> bf16, packed two per dword
static __device__ __forceinline__ unsigned int pack_bf16x2(float a, float b) {
#if __has_builtin(__builtin_amdgcn_cvt_pk_bf16_f32)
    v2bf p = __builtin_amdgcn_cvt_pk_bf16_f32(a, b);
    return __builtin_bit_cast(unsigned int, p);
#else
    unsigned int ua = __float_as_uint(a);
    ua = (ua + 0x7FFFu + ((ua >> 16) & 1u)) >> 16;
    unsigned int ub = __float_as_uint(b);
    ub = (ub + 0x7FFFu + ((ub >> 16) & 1u)) >> 16;
    return (ua & 0xFFFFu) | (ub << 16);
#endif
}

// ---- CDNA5 async global->LDS copy (ASYNCcnt-tracked, no VGPR round trip) ---
// ISA 10.7 / 15.18.3: GLOBAL_LOAD_ASYNC_TO_LDS_B128, GV mode.
// VDST = LDS byte address, VADDR = 64-bit global address.
static __device__ __forceinline__ void async_copy_b128(unsigned lds_addr, const void* gaddr) {
    asm volatile("global_load_async_to_lds_b128 %0, %1, off"
                 :: "v"(lds_addr), "v"((unsigned long long)(uintptr_t)gaddr)
                 : "memory");
}
static __device__ __forceinline__ void wait_async0() {
    asm volatile("s_wait_asynccnt 0x0" ::: "memory");
}
static __device__ __forceinline__ unsigned lds_addr_of(const void* p) {
    // generic LDS pointers carry the LDS offset in the low 32 bits
    return (unsigned)(uintptr_t)p;
}

// A fragment (16x32 bf16, ISA 7.12.2 layout):
//  lanes 0-15 : M=lane,   elems 0-7 = K 0-7,  elems 8-15 = K 16-23
//  lanes16-31 : M=lane-16,elems 0-7 = K 8-15, elems 8-15 = K 24-31
// LDS tile stored row-major [M][32 bf16] = [M][16 dwords]
static __device__ __forceinline__ v16bf frag_a(const unsigned int* tile, int mOff, int lane) {
    const int m  = mOff + (lane & 15);
    const int i0 = m * 16 + ((lane >> 4) & 1) * 4;  // dword index (16B chunk)
    V4x2 t;
    t.a = *(const v4u*)(tile + i0);      // K 0-7  (or 8-15)
    t.b = *(const v4u*)(tile + i0 + 8);  // K 16-23 (or 24-31)
    return __builtin_bit_cast(v16bf, t);
}

// B fragment (32x16 bf16): lane L holds column N=L%16; elems 0-7 = K 8h..8h+7, elems 8-15 = K 16+8h..23+8h (h=L/16).
// LDS tile stored row-major [K=32][64 bf16] = [K][32 dwords]
static __device__ __forceinline__ v16bf frag_b(const unsigned int* tile, int nOffDwords, int lane) {
    const unsigned short* tb = (const unsigned short*)tile;
    const int n = nOffDwords * 2 + (lane & 15);
    const int h = (lane >> 4) & 1;
    v16bf r;
    #pragma unroll
    for (int s = 0; s < 16; ++s) {
        const int k = (s < 8) ? (8 * h + s) : (16 + 8 * h + (s - 8));
        r[s] = __builtin_bit_cast(__bf16, tb[k * 64 + n]);
    }
    return r;
}

// shared inner step: 12x v_wmma_f32_16x16x32_bf16 per wave
static __device__ __forceinline__ void mma_step(
    const unsigned int* axh, const unsigned int* ax2h,
    const unsigned int* bwr, const unsigned int* bwg, const unsigned int* bwb,
    int lane, int wm, int wn,
    v8f (&accR)[2][2], v8f (&accG)[2][2], v8f (&accB)[2][2])
{
    v16bf af[2], a2f[2], rf[2], gf[2], bf_[2];
    #pragma unroll
    for (int tm = 0; tm < 2; ++tm) {
        af [tm] = frag_a(axh,  wm * 32 + tm * 16, lane);
        a2f[tm] = frag_a(ax2h, wm * 32 + tm * 16, lane);
    }
    #pragma unroll
    for (int tn = 0; tn < 2; ++tn) {
        const int nOff = wn * 16 + tn * 8;
        rf [tn] = frag_b(bwr, nOff, lane);
        gf [tn] = frag_b(bwg, nOff, lane);
        bf_[tn] = frag_b(bwb, nOff, lane);
    }
    #pragma unroll
    for (int tm = 0; tm < 2; ++tm)
        #pragma unroll
        for (int tn = 0; tn < 2; ++tn) {
            accR[tm][tn] = __builtin_amdgcn_wmma_f32_16x16x32_bf16(
                false, af[tm],  false, rf[tn],  (short)0, accR[tm][tn], false, false);
            accG[tm][tn] = __builtin_amdgcn_wmma_f32_16x16x32_bf16(
                false, af[tm],  false, gf[tn],  (short)0, accG[tm][tn], false, false);
            accB[tm][tn] = __builtin_amdgcn_wmma_f32_16x16x32_bf16(
                false, a2f[tm], false, bf_[tn], (short)0, accB[tm][tn], false, false);
        }
}

// shared epilogue: (zr*zg + zb) * percent[o]
// C/D layout: VGPR r -> M = r + (lane>=16 ? 8 : 0), N = lane&15
static __device__ __forceinline__ void epilogue(
    const float* br, const float* bg, const float* bb, const float* pct,
    float* out, int mBase, int nBase, int lane, int wm, int wn,
    v8f (&accR)[2][2], v8f (&accG)[2][2], v8f (&accB)[2][2])
{
    #pragma unroll
    for (int tn = 0; tn < 2; ++tn) {
        const int o = nBase + wn * 32 + tn * 16 + (lane & 15);
        const float vbr = br[o], vbg = bg[o], vbb = bb[o], vp = pct[o];
        #pragma unroll
        for (int tm = 0; tm < 2; ++tm) {
            const int mrow0 = mBase + wm * 32 + tm * 16 + ((lane >> 4) << 3);
            #pragma unroll
            for (int r = 0; r < 8; ++r) {
                const float zr = accR[tm][tn][r] + vbr;
                const float zg = accG[tm][tn][r] + vbg;
                const float zb = accB[tm][tn][r] + vbb;
                out[(size_t)(mrow0 + r) * OUTS + o] = (zr * zg + zb) * vp;
            }
        }
    }
}

// ---------------------------------------------------------------------------
// Kernel 1: percent[o] = sigmoid(segment_out[o, argmax_s |segment_out[o,s]|])
// ---------------------------------------------------------------------------
__global__ __launch_bounds__(64) void seg_gate_kernel(
    const float* __restrict__ ctx, const float* __restrict__ segW,
    const float* __restrict__ segB, float* __restrict__ percent)
{
    __shared__ float sm[64];
    const int d = blockIdx.x;
    const int t = threadIdx.x;
    const int s = t & 15;
    const int chunk = t >> 4;

    const float* wp = segW + (size_t)d * DCTX * NSEG + s;
    float acc = 0.f;
    const int c0 = chunk * (DCTX / 4);
    #pragma unroll 4
    for (int c = c0; c < c0 + DCTX / 4; ++c)
        acc += ctx[c] * wp[(size_t)c * NSEG];

    sm[t] = acc;
    __syncthreads();
    if (t < 16) {
        float v = sm[s] + sm[16 + s] + sm[32 + s] + sm[48 + s] + segB[d * NSEG + s];
        sm[s] = v;
    }
    __syncthreads();
    if (t == 0) {
        float best = sm[0], ba = fabsf(sm[0]);
        #pragma unroll
        for (int i = 1; i < NSEG; ++i) {
            float a = fabsf(sm[i]);
            if (a > ba) { ba = a; best = sm[i]; }   // strict '>' == first-max (jnp.argmax)
        }
        percent[d] = 1.f / (1.f + __expf(-best));
    }
}

// ---------------------------------------------------------------------------
// Conversion pre-kernels (bandwidth-bound, one-time): f32 -> bf16 mirrors.
// ---------------------------------------------------------------------------
__global__ __launch_bounds__(256) void convert_x_kernel(
    const float* __restrict__ x, unsigned int* __restrict__ xh,
    unsigned int* __restrict__ x2h)
{
    const size_t i = ((size_t)blockIdx.x * 256 + threadIdx.x) * 4;  // float index
    const float4 v = *(const float4*)(x + i);
    uint2 p, q;
    p.x = pack_bf16x2(v.x, v.y);           p.y = pack_bf16x2(v.z, v.w);
    q.x = pack_bf16x2(v.x * v.x, v.y * v.y); q.y = pack_bf16x2(v.z * v.z, v.w * v.w);
    *(uint2*)(xh  + i / 2) = p;
    *(uint2*)(x2h + i / 2) = q;
}

__global__ __launch_bounds__(256) void convert_w_kernel(
    const float* __restrict__ w, unsigned int* __restrict__ wh)
{
    const size_t i = ((size_t)blockIdx.x * 256 + threadIdx.x) * 4;
    const float4 v = *(const float4*)(w + i);
    uint2 p;
    p.x = pack_bf16x2(v.x, v.y); p.y = pack_bf16x2(v.z, v.w);
    *(uint2*)(wh + i / 2) = p;
}

// ---------------------------------------------------------------------------
// Kernel 2a (fast path): fused triple GEMM from bf16 mirrors, tiles staged
// into LDS with GLOBAL_LOAD_ASYNC_TO_LDS_B128 (ASYNCcnt), zero hot-loop VALU
// for conversion. Block 256 thr = 8 wave32, tile 128(M) x 64(N).
// ---------------------------------------------------------------------------
__global__ __launch_bounds__(256) void qad_gemm_bf16_kernel(
    const unsigned short* __restrict__ xh, const unsigned short* __restrict__ x2h,
    const unsigned short* __restrict__ wrh, const unsigned short* __restrict__ wgh,
    const unsigned short* __restrict__ wbh,
    const float* __restrict__ br, const float* __restrict__ bg, const float* __restrict__ bb,
    const float* __restrict__ pct, float* __restrict__ out)
{
    __shared__ unsigned int axh [128 * 16];   // x   tile 128x32 bf16 (8 KB)
    __shared__ unsigned int ax2h[128 * 16];   // x^2 tile 128x32 bf16 (8 KB)
    __shared__ unsigned int bwr [32 * 32];    // W tiles 32x64 bf16 (4 KB each)
    __shared__ unsigned int bwg [32 * 32];
    __shared__ unsigned int bwb [32 * 32];

    const int t    = threadIdx.x;
    const int lane = t & 31;
    const int wave = t >> 5;
    const int wm   = wave & 3;
    const int wn   = wave >> 2;
    const int mBase = blockIdx.y * 128;
    const int nBase = blockIdx.x * 64;

    v8f accR[2][2], accG[2][2], accB[2][2];
    #pragma unroll
    for (int i = 0; i < 2; ++i)
        #pragma unroll
        for (int j = 0; j < 2; ++j) {
            accR[i][j] = (v8f)0.f; accG[i][j] = (v8f)0.f; accB[i][j] = (v8f)0.f;
        }

    // A staging: 128 rows x 64B = 512 x b128; 2 per thread
    const int aIdx0 = t * 2;
    // W staging: 32 rows x 128B = 256 x b128; 1 per thread per matrix
    const int wRow = t >> 3;          // 0..31
    const int wCh  = (t & 7) * 16;    // byte chunk in 128B row

    const unsigned axBase  = lds_addr_of(axh);
    const unsigned ax2Base = lds_addr_of(ax2h);
    const unsigned wrBase  = lds_addr_of(bwr);
    const unsigned wgBase  = lds_addr_of(bwg);
    const unsigned wbBase  = lds_addr_of(bwb);

    for (int k0 = 0; k0 < INSZ; k0 += 32) {
        // ---- async-stage A / A^2 tiles (bf16 rows of 64B) ----
        #pragma unroll
        for (int j = 0; j < 2; ++j) {
            const int idx = aIdx0 + j;          // 0..511
            const int row = idx >> 2;           // 0..127
            const int ch  = (idx & 3) * 16;     // byte chunk in 64B row
            const size_t gOff = ((size_t)(mBase + row) * INSZ + k0) * 2 + ch;
            async_copy_b128(axBase  + row * 64 + ch, (const char*)xh  + gOff);
            async_copy_b128(ax2Base + row * 64 + ch, (const char*)x2h + gOff);
        }
        // ---- async-stage the three weight tiles (bf16 rows of 128B) ----
        {
            const size_t gOff = ((size_t)(k0 + wRow) * OUTS + nBase) * 2 + wCh;
            const unsigned l = wRow * 128 + wCh;
            async_copy_b128(wrBase + l, (const char*)wrh + gOff);
            async_copy_b128(wgBase + l, (const char*)wgh + gOff);
            async_copy_b128(wbBase + l, (const char*)wbh + gOff);
        }
        wait_async0();
        __syncthreads();

        mma_step(axh, ax2h, bwr, bwg, bwb, lane, wm, wn, accR, accG, accB);

        __syncthreads();
    }

    epilogue(br, bg, bb, pct, out, mBase, nBase, lane, wm, wn, accR, accG, accB);
}

// ---------------------------------------------------------------------------
// Kernel 2b (fallback, workspace too small): round-1 fused-convert GEMM.
// ---------------------------------------------------------------------------
__global__ __launch_bounds__(256) void qad_gemm_f32_kernel(
    const float* __restrict__ x,
    const float* __restrict__ Wr, const float* __restrict__ Wg, const float* __restrict__ Wb,
    const float* __restrict__ br, const float* __restrict__ bg, const float* __restrict__ bb,
    const float* __restrict__ pct, float* __restrict__ out)
{
    __shared__ unsigned int axh [128 * 16];
    __shared__ unsigned int ax2h[128 * 16];
    __shared__ unsigned int bwr [32 * 32];
    __shared__ unsigned int bwg [32 * 32];
    __shared__ unsigned int bwb [32 * 32];

    const int t    = threadIdx.x;
    const int lane = t & 31;
    const int wave = t >> 5;
    const int wm   = wave & 3;
    const int wn   = wave >> 2;
    const int mBase = blockIdx.y * 128;
    const int nBase = blockIdx.x * 64;

    v8f accR[2][2], accG[2][2], accB[2][2];
    #pragma unroll
    for (int i = 0; i < 2; ++i)
        #pragma unroll
        for (int j = 0; j < 2; ++j) {
            accR[i][j] = (v8f)0.f; accG[i][j] = (v8f)0.f; accB[i][j] = (v8f)0.f;
        }

    const int xr  = t >> 3;
    const int xc  = (t & 7) * 4;
    const int wk  = t >> 4;
    const int wn4 = (t & 15) * 4;

    for (int k0 = 0; k0 < INSZ; k0 += 32) {
        #pragma unroll
        for (int i = 0; i < 4; ++i) {
            const int m = xr + i * 32;
            const float4 v = *(const float4*)(x + (size_t)(mBase + m) * INSZ + k0 + xc);
            const int d0 = m * 16 + (xc >> 1);
            axh [d0]     = pack_bf16x2(v.x, v.y);
            axh [d0 + 1] = pack_bf16x2(v.z, v.w);
            ax2h[d0]     = pack_bf16x2(v.x * v.x, v.y * v.y);
            ax2h[d0 + 1] = pack_bf16x2(v.z * v.z, v.w * v.w);
        }
        #pragma unroll
        for (int j = 0; j < 2; ++j) {
            const int k = wk + j * 16;
            const size_t g = (size_t)(k0 + k) * OUTS + nBase + wn4;
            const int d0 = k * 32 + (wn4 >> 1);
            const float4 vr = *(const float4*)(Wr + g);
            bwr[d0] = pack_bf16x2(vr.x, vr.y); bwr[d0 + 1] = pack_bf16x2(vr.z, vr.w);
            const float4 vg = *(const float4*)(Wg + g);
            bwg[d0] = pack_bf16x2(vg.x, vg.y); bwg[d0 + 1] = pack_bf16x2(vg.z, vg.w);
            const float4 vb = *(const float4*)(Wb + g);
            bwb[d0] = pack_bf16x2(vb.x, vb.y); bwb[d0 + 1] = pack_bf16x2(vb.z, vb.w);
        }
        if (k0 + 32 < INSZ)
            __builtin_prefetch(x + (size_t)(mBase + xr) * INSZ + k0 + 32 + xc, 0, 0);

        __syncthreads();
        mma_step(axh, ax2h, bwr, bwg, bwb, lane, wm, wn, accR, accG, accB);
        __syncthreads();
    }

    epilogue(br, bg, bb, pct, out, mBase, nBase, lane, wm, wn, accR, accG, accB);
}

// ---------------------------------------------------------------------------
extern "C" void kernel_launch(void* const* d_in, const int* in_sizes, int n_in,
                              void* d_out, int out_size, void* d_ws, size_t ws_size,
                              hipStream_t stream)
{
    const float* x    = (const float*)d_in[0];
    const float* ctx  = (const float*)d_in[1];
    const float* Wr   = (const float*)d_in[2];
    const float* Wg   = (const float*)d_in[3];
    const float* Wb   = (const float*)d_in[4];
    const float* br   = (const float*)d_in[5];
    const float* bg   = (const float*)d_in[6];
    const float* bb   = (const float*)d_in[7];
    const float* segW = (const float*)d_in[8];
    const float* segB = (const float*)d_in[9];
    float* out = (float*)d_out;

    // bf16-mirror workspace layout
    const size_t xhB  = (size_t)BSZ * INSZ * 2;    // 32 MB
    const size_t whB  = (size_t)INSZ * OUTS * 2;   //  8 MB each
    const size_t need = 2 * xhB + 3 * whB + (size_t)OUTS * 4;

    dim3 grid(OUTS / 64, BSZ / 128);

    if (ws_size >= need) {
        char* ws   = (char*)d_ws;
        unsigned int* xh  = (unsigned int*)(ws);
        unsigned int* x2h = (unsigned int*)(ws + xhB);
        unsigned int* wrh = (unsigned int*)(ws + 2 * xhB);
        unsigned int* wgh = (unsigned int*)(ws + 2 * xhB + whB);
        unsigned int* wbh = (unsigned int*)(ws + 2 * xhB + 2 * whB);
        float*        pct = (float*)      (ws + 2 * xhB + 3 * whB);

        seg_gate_kernel<<<dim3(OUTS), dim3(64), 0, stream>>>(ctx, segW, segB, pct);
        convert_x_kernel<<<dim3(BSZ * INSZ / 1024), dim3(256), 0, stream>>>(x, xh, x2h);
        convert_w_kernel<<<dim3(INSZ * OUTS / 1024), dim3(256), 0, stream>>>(Wr, wrh);
        convert_w_kernel<<<dim3(INSZ * OUTS / 1024), dim3(256), 0, stream>>>(Wg, wgh);
        convert_w_kernel<<<dim3(INSZ * OUTS / 1024), dim3(256), 0, stream>>>(Wb, wbh);

        qad_gemm_bf16_kernel<<<grid, dim3(256), 0, stream>>>(
            (const unsigned short*)xh, (const unsigned short*)x2h,
            (const unsigned short*)wrh, (const unsigned short*)wgh,
            (const unsigned short*)wbh, br, bg, bb, pct, out);
    } else {
        float* pct = (float*)d_ws;   // only OUTS floats needed
        seg_gate_kernel<<<dim3(OUTS), dim3(64), 0, stream>>>(ctx, segW, segB, pct);
        qad_gemm_f32_kernel<<<grid, dim3(256), 0, stream>>>(
            x, Wr, Wg, Wb, br, bg, bb, pct, out);
    }
}
